// CenterAttention3D_7086696039096
// MI455X (gfx1250) — compile-verified, matrix operand order checked by reading, not run on hardware
//
#include <hip/hip_runtime.h>
#include <hip/hip_bf16.h>

// ---------------------------------------------------------------------------
// CenterAttention3D on MI455X (gfx1250): bf16 WMMA for the three GEMMs
// (2x2 register-blocked: one wave -> 32x32 output block, 24 static WMMAs),
// VALU for the irregular 27-neighbor windowed softmax.
// ---------------------------------------------------------------------------

typedef __attribute__((ext_vector_type(16))) __bf16 v16bf;
typedef __attribute__((ext_vector_type(8)))  float  v8f;

#define C_DIM   192
#define QKV_DIM 576
#define HC      48
#define NHEAD   4
#define KWIN    27

__device__ __forceinline__ unsigned short f2bf(float f) {
    unsigned int u = __float_as_uint(f);
    // round-to-nearest-even truncation to bf16
    u = (u + 0x7FFFu + ((u >> 16) & 1u)) >> 16;
    return (unsigned short)u;
}

// ---------------- pack X (f32 -> bf16) ----------------
__global__ void pack_x_kernel(const float* __restrict__ x,
                              unsigned short* __restrict__ xbf, int n) {
    int i = blockIdx.x * blockDim.x + threadIdx.x;
    if (i < n) xbf[i] = f2bf(x[i]);
}

// ---------------- pack weights, transposed (Bt[n][k]) ----------------
// Wt  (576 x 192): cols 0..191 from Wq, cols 192..575 from Wkv (k then v)
// Wpt (192 x 192): from Wp
__global__ void pack_w_kernel(const float* __restrict__ Wq,
                              const float* __restrict__ Wkv,
                              const float* __restrict__ Wp,
                              unsigned short* __restrict__ Wt,
                              unsigned short* __restrict__ Wpt) {
    int i = blockIdx.x * blockDim.x + threadIdx.x;
    const int n_wt = QKV_DIM * C_DIM;
    const int n_wp = C_DIM * C_DIM;
    if (i < n_wt) {
        int n = i / C_DIM, k = i % C_DIM;
        float v = (n < C_DIM) ? Wq[k * C_DIM + n]
                              : Wkv[k * (2 * C_DIM) + (n - C_DIM)];
        Wt[i] = f2bf(v);
    } else if (i < n_wt + n_wp) {
        int j = i - n_wt;
        int n = j / C_DIM, k = j % C_DIM;
        Wpt[j] = f2bf(Wp[k * C_DIM + n]);
    }
}

// ---------------- bf16 WMMA GEMM, 2x2 register-blocked ----------------------
// A:  M x K bf16 row-major. Bt: Ncols x K bf16 row-major (i.e. B transposed).
// One wave computes a 32x32 output block (2x2 tiles of 16x16).
// K == 192 (6 steps of 32); M multiple of 32; Ncols multiple of 32.
// Per-lane fragment layout (ISA 7.12.2, 16-bit A 16x32): lane m = lane&15,
// K elements {base..base+7, base+16..base+23}, base = 8*(lane>=16).
__global__ __launch_bounds__(256) void wmma_gemm_kernel(
    const unsigned short* __restrict__ A,
    const unsigned short* __restrict__ Bt,
    const float* __restrict__ bias0,
    const float* __restrict__ bias1, int split,
    float* __restrict__ Cout,
    int M, int K, int Ncols, int scale_cols, float scale) {

    int wave = (int)((blockIdx.x * blockDim.x + threadIdx.x) >> 5);
    int lane = threadIdx.x & 31;
    int blocks_n = Ncols >> 5;                // 32-wide blocks
    int bm = wave / blocks_n;
    int bn = wave - bm * blocks_n;
    if (bm * 32 >= M) return;                 // wave-uniform: EXEC stays all-1s

    int r16   = lane & 15;
    int hi    = lane >> 4;
    int kbase = hi << 3;                      // 0 or 8

    const unsigned short* arow0 = A  + (size_t)(bm * 32 + r16)      * K;
    const unsigned short* arow1 = A  + (size_t)(bm * 32 + 16 + r16) * K;
    const unsigned short* brow0 = Bt + (size_t)(bn * 32 + r16)      * K;
    const unsigned short* brow1 = Bt + (size_t)(bn * 32 + 16 + r16) * K;

    v8f acc00 = {}, acc01 = {}, acc10 = {}, acc11 = {};
#pragma unroll
    for (int kk = 0; kk < 192; kk += 32) {    // K == 192 for all three GEMMs
        union { uint4 q[2]; v16bf v; } a0, a1, b0, b1;
        a0.q[0] = *(const uint4*)(arow0 + kk + kbase);
        a0.q[1] = *(const uint4*)(arow0 + kk + kbase + 16);
        a1.q[0] = *(const uint4*)(arow1 + kk + kbase);
        a1.q[1] = *(const uint4*)(arow1 + kk + kbase + 16);
        b0.q[0] = *(const uint4*)(brow0 + kk + kbase);
        b0.q[1] = *(const uint4*)(brow0 + kk + kbase + 16);
        b1.q[0] = *(const uint4*)(brow1 + kk + kbase);
        b1.q[1] = *(const uint4*)(brow1 + kk + kbase + 16);
        acc00 = __builtin_amdgcn_wmma_f32_16x16x32_bf16(false, a0.v, false, b0.v,
                                                        (short)0, acc00, false, false);
        acc01 = __builtin_amdgcn_wmma_f32_16x16x32_bf16(false, a0.v, false, b1.v,
                                                        (short)0, acc01, false, false);
        acc10 = __builtin_amdgcn_wmma_f32_16x16x32_bf16(false, a1.v, false, b0.v,
                                                        (short)0, acc10, false, false);
        acc11 = __builtin_amdgcn_wmma_f32_16x16x32_bf16(false, a1.v, false, b1.v,
                                                        (short)0, acc11, false, false);
    }

    // Epilogue. D layout: element r -> M = tile_m0 + r + 8*hi, N = tile_n0 + (lane&15).
    int n0 = bn * 32 + r16;
    int n1 = n0 + 16;
    float bA = (n0 < split) ? bias0[n0] : bias1[n0 - split];
    float bB = (n1 < split) ? bias0[n1] : bias1[n1 - split];
    float sA = (n0 < scale_cols) ? scale : 1.0f;
    float sB = (n1 < scale_cols) ? scale : 1.0f;
    int m0 = bm * 32 + hi * 8;                // rows for acc0x
    int m1 = m0 + 16;                         // rows for acc1x
#pragma unroll
    for (int r = 0; r < 8; ++r) {
        Cout[(size_t)(m0 + r) * Ncols + n0] = (acc00[r] + bA) * sA;
        Cout[(size_t)(m0 + r) * Ncols + n1] = (acc01[r] + bB) * sB;
        Cout[(size_t)(m1 + r) * Ncols + n0] = (acc10[r] + bA) * sA;
        Cout[(size_t)(m1 + r) * Ncols + n1] = (acc11[r] + bB) * sB;
    }
}

// ---------------- attention pass 1: softmax weights per (voxel, head) -------
// Zero-padded semantics: OOB neighbor has logit 0 and stays in the softmax.
__global__ void attn_logits_kernel(const float* __restrict__ qkv,
                                   float* __restrict__ wgt,
                                   int D, int H, int W) {
    int idx = blockIdx.x * blockDim.x + threadIdx.x;
    int N = D * H * W;
    if (idx >= N * NHEAD) return;
    int v = idx >> 2, h = idx & 3;
    int x = v % W, y = (v / W) % H, z = v / (W * H);

    float qreg[HC];
    const float* qp = qkv + (size_t)v * QKV_DIM + h * HC;
#pragma unroll
    for (int c = 0; c < HC; ++c) qreg[c] = qp[c];

    float logit[KWIN];
#pragma unroll 1
    for (int j = 0; j < KWIN; ++j) {
        int dz = j / 9 - 1, dy = (j / 3) % 3 - 1, dx = j % 3 - 1;
        int zz = z + dz, yy = y + dy, xx = x + dx;
        float acc = 0.0f;
        if (zz >= 0 && zz < D && yy >= 0 && yy < H && xx >= 0 && xx < W) {
            const float* kp = qkv + (size_t)((zz * H + yy) * W + xx) * QKV_DIM
                            + C_DIM + h * HC;
#pragma unroll
            for (int c = 0; c < HC; ++c) acc += qreg[c] * kp[c];
        }
        logit[j] = acc;
    }

    float mx = logit[0];
#pragma unroll
    for (int j = 1; j < KWIN; ++j) mx = fmaxf(mx, logit[j]);
    float sum = 0.0f;
#pragma unroll
    for (int j = 0; j < KWIN; ++j) { float e = __expf(logit[j] - mx); logit[j] = e; sum += e; }
    float inv = 1.0f / sum;
    float* wp = wgt + (size_t)idx * KWIN;
#pragma unroll
    for (int j = 0; j < KWIN; ++j) wp[j] = logit[j] * inv;
}

// ---------------- attention pass 2: out[v,c] = sum_j w[v,h,j] * V[nbr,c] ----
__global__ void attn_apply_kernel(const float* __restrict__ qkv,
                                  const float* __restrict__ wgt,
                                  unsigned short* __restrict__ outbf,
                                  int D, int H, int W) {
    int idx = blockIdx.x * blockDim.x + threadIdx.x;
    int N = D * H * W;
    if (idx >= N * C_DIM) return;
    int v = idx / C_DIM, c = idx % C_DIM;
    int h = c / HC;
    int x = v % W, y = (v / W) % H, z = v / (W * H);
    const float* wp = wgt + (size_t)(v * NHEAD + h) * KWIN;

    float acc = 0.0f;
#pragma unroll 1
    for (int j = 0; j < KWIN; ++j) {
        int dz = j / 9 - 1, dy = (j / 3) % 3 - 1, dx = j % 3 - 1;
        int zz = z + dz, yy = y + dy, xx = x + dx;
        if (zz >= 0 && zz < D && yy >= 0 && yy < H && xx >= 0 && xx < W) {
            acc += wp[j] * qkv[(size_t)((zz * H + yy) * W + xx) * QKV_DIM
                               + 2 * C_DIM + c];
        }
    }
    outbf[idx] = f2bf(acc);
}

// ---------------------------------------------------------------------------
static inline size_t align256(size_t x) { return (x + 255) & ~(size_t)255; }

extern "C" void kernel_launch(void* const* d_in, const int* in_sizes, int n_in,
                              void* d_out, int out_size, void* d_ws, size_t ws_size,
                              hipStream_t stream) {
    const float* x   = (const float*)d_in[0];
    const float* Wq  = (const float*)d_in[1];
    const float* bq  = (const float*)d_in[2];
    const float* Wkv = (const float*)d_in[3];
    const float* bkv = (const float*)d_in[4];
    const float* Wp  = (const float*)d_in[5];
    const float* bp  = (const float*)d_in[6];
    float* out = (float*)d_out;

    const int N = in_sizes[0] / C_DIM;        // 21952
    int dim = 1; while (dim * dim * dim < N) ++dim;  // 28
    const int D = dim, H = dim, W = dim;
    const float scale = 0.14433756729740643f; // 48^-0.5

    // workspace layout
    char* ws = (char*)d_ws;
    size_t off = 0;
    unsigned short* xbf = (unsigned short*)(ws + off); off = align256(off + (size_t)N * C_DIM * 2);
    unsigned short* Wt  = (unsigned short*)(ws + off); off = align256(off + (size_t)QKV_DIM * C_DIM * 2);
    unsigned short* Wpt = (unsigned short*)(ws + off); off = align256(off + (size_t)C_DIM * C_DIM * 2);
    float* qkv          = (float*)(ws + off);          off = align256(off + (size_t)N * QKV_DIM * 4);
    float* attw         = (float*)(ws + off);          off = align256(off + (size_t)N * NHEAD * KWIN * 4);
    unsigned short* abf = (unsigned short*)(ws + off); off = align256(off + (size_t)N * C_DIM * 2);

    // 1) pack inputs to bf16 (weights pre-transposed)
    {
        int n = N * C_DIM;
        pack_x_kernel<<<(n + 255) / 256, 256, 0, stream>>>(x, xbf, n);
        int nw = QKV_DIM * C_DIM + C_DIM * C_DIM;
        pack_w_kernel<<<(nw + 255) / 256, 256, 0, stream>>>(Wq, Wkv, Wp, Wt, Wpt);
    }

    // 2) fused QKV GEMM: qkv = X @ [Wq|Wkv] + [bq|bkv], q-part scaled
    {
        int blocks32 = (N / 32) * (QKV_DIM / 32);   // waves (32x32 blocks)
        int blocks = (blocks32 + 7) / 8;            // 8 waves / workgroup
        wmma_gemm_kernel<<<blocks, 256, 0, stream>>>(
            xbf, Wt, bq, bkv, /*split=*/C_DIM, qkv,
            N, C_DIM, QKV_DIM, /*scale_cols=*/C_DIM, scale);
    }

    // 3) windowed softmax weights, then weighted value sum (bf16 out)
    {
        int n1 = N * NHEAD;
        attn_logits_kernel<<<(n1 + 255) / 256, 256, 0, stream>>>(qkv, attw, D, H, W);
        int n2 = N * C_DIM;
        attn_apply_kernel<<<(n2 + 255) / 256, 256, 0, stream>>>(qkv, attw, abf, D, H, W);
    }

    // 4) output projection GEMM: out = Attn @ Wp + bp
    {
        int blocks32 = (N / 32) * (C_DIM / 32);
        int blocks = (blocks32 + 7) / 8;
        wmma_gemm_kernel<<<blocks, 256, 0, stream>>>(
            abf, Wpt, bp, bp, /*split=*/C_DIM, out,
            N, C_DIM, C_DIM, /*scale_cols=*/0, 1.0f);
    }
}
